// Point_TemporalPriorEncoder_6047313953109
// MI455X (gfx1250) — matrix operand order repaired and checked by reading, not compiled, not run on hardware
//
#include <hip/hip_runtime.h>

// ---------------------------------------------------------------------------
// CDNA5 (gfx1250) sparse-conv network via V_WMMA_F32_16X16X32_F16, wave32.
// Every stage: prefetch neighbor indices (registers) -> stage weight chunk in
// LDS (DS pipe) -> gather activation rows (VMEM, L2-resident) -> 16x16x32
// WMMA -> bias/relu/residual -> store. f16 activations, f32 accumulation.
// All tap counts / chunking compile-time so index prefetch + compute fully
// unroll and stay register-resident.
// ---------------------------------------------------------------------------

typedef __attribute__((ext_vector_type(16))) _Float16 v16h;
typedef __attribute__((ext_vector_type(8)))  _Float16 v8h;
typedef __attribute__((ext_vector_type(8)))  float    v8f;

#define N_PTS_C 64
#define LDS_CAP_HALFS 32768   // 64 KB weight staging cap per workgroup

// Convert f32 weights [K, Cin, Cout] -> f16 transposed [K, Cout, CinPad]
// (pad columns CinPad>Cin and pad K-slices stay zero from workspace memset).
__global__ __launch_bounds__(256) void wconv_kernel(
    const float* __restrict__ W, _Float16* __restrict__ WT,
    int Cin, int Cout, int CinPad, int total)
{
    int idx = blockIdx.x * blockDim.x + threadIdx.x;
    if (idx >= total) return;
    int co = idx % Cout;
    int t  = idx / Cout;
    int ci = t % Cin;
    int k  = t / Cin;
    WT[((size_t)(k * Cout + co)) * CinPad + ci] = (_Float16)W[idx];
}

// Plain f32 -> f16 copy (feature matrix); padding row stays zero from memset.
__global__ __launch_bounds__(256) void fconv_kernel(
    const float* __restrict__ s, _Float16* __restrict__ d, int n)
{
    int i = blockIdx.x * blockDim.x + threadIdx.x;
    if (i < n) d[i] = (_Float16)s[i];
}

// Gathered sparse-conv GEMM stage.
//   src    : f16, rows padded to (padRow+1); padding row index appears in nbr.
//   nbr    : [nOut, KOFF] int32 (nullptr => identity map, KOFF must be 1)
//   WT     : f16 [KOFFPAD, COUT, CINPAD] transposed weights (zero K-pad slices)
//   NC     : 16-wide output column chunks (cout = 16*NC)
//   KSTEPS : Cin/32 contraction steps (cinPad = 32*KSTEPS)
//   KOFF   : real tap count; KCH taps staged in LDS per chunk.
// One wave per 16-row output tile; 8 waves (8 tiles) per 256-thread block.
template<int NC, int KSTEPS, int KOFF, int KCH>
__global__ __launch_bounds__(256) void sconv16_kernel(
    const _Float16* __restrict__ src, int srcStride,
    const int* __restrict__ nbr,
    const _Float16* __restrict__ WT,
    const float* __restrict__ bias, int nOut, int padRow,
    _Float16* __restrict__ dstH, float* __restrict__ dstF,
    int dstStride, int dstColOff,
    const _Float16* __restrict__ resid,
    int doRelu, int padZero)
{
    constexpr int COUT    = 16 * NC;
    constexpr int CINPAD  = 32 * KSTEPS;
    constexpr int WSLICE  = COUT * CINPAD;                 // halfs per tap
    constexpr int KOFFPAD = ((KOFF + KCH - 1) / KCH) * KCH;

    extern __shared__ _Float16 sW[];

    const int lane  = threadIdx.x & 31;
    const int wave  = threadIdx.x >> 5;
    const int nTile = (nOut + 15) >> 4;
    // No early return: every wave must participate in LDS staging. Overflow
    // waves clamp to the last tile and recompute identical values; their
    // stores duplicate the owner wave's bytes (benign, deterministic).
    int tile = blockIdx.x * (blockDim.x >> 5) + wave;
    if (tile > nTile - 1) tile = nTile - 1;
    const int outBase = tile << 4;
    const int l  = lane & 15;
    const int hi = lane >> 4;
    int orow = outBase + l;
    if (orow > nOut - 1) orow = nOut - 1;      // tail clamp (stores guarded)

    v8f acc[NC];
#pragma unroll
    for (int nc = 0; nc < NC; ++nc) {
        float bv = bias ? bias[nc * 16 + l] : 0.0f;
#pragma unroll
        for (int r = 0; r < 8; ++r) acc[nc][r] = bv;
    }

    for (int kc = 0; kc < KOFFPAD; kc += KCH) {
        // Prefetch this chunk's gather indices into registers; these loads
        // overlap the LDS staging below. Phantom taps (k >= KOFF) map to the
        // zero padding row and zero weight slices -> contribute exactly 0.
        int rows[KCH];
#pragma unroll
        for (int kk = 0; kk < KCH; ++kk) {
            const int k = kc + kk;
            rows[kk] = (k < KOFF)
                           ? (nbr ? nbr[(size_t)orow * KOFF + k] : orow)
                           : padRow;
        }

        __syncthreads();                       // previous chunk fully consumed
#pragma unroll
        for (int i = 0; i < KCH * WSLICE; i += 256 * 8) {
            const int e = i + threadIdx.x * 8;
            if (KCH * WSLICE % (256 * 8) == 0 || e < KCH * WSLICE)
                *reinterpret_cast<v8h*>(sW + e) =
                    *reinterpret_cast<const v8h*>(WT + (size_t)kc * WSLICE + e);
        }
        __syncthreads();                       // chunk visible to all waves

#pragma unroll
        for (int kk = 0; kk < KCH; ++kk) {
            const _Float16* srow = src + (size_t)rows[kk] * srcStride;
#pragma unroll
            for (int ks = 0; ks < KSTEPS; ++ks) {
                const int kb = ks * 32;
                // A fragment, 16-bit 16x32 layout: lanes 0-15 hold
                // K[0..7]+K[16..23], lanes 16-31 hold K[8..15]+K[24..31],
                // for row M = lane%16.
                v8h a0 = *reinterpret_cast<const v8h*>(srow + kb + hi * 8);
                v8h a1 = *reinterpret_cast<const v8h*>(srow + kb + 16 + hi * 8);
                v16h a;
#pragma unroll
                for (int j = 0; j < 8; ++j) { a[j] = a0[j]; a[8 + j] = a1[j]; }
#pragma unroll
                for (int nc = 0; nc < NC; ++nc) {
                    // B fragment, 32x16: lane holds column N=lane%16,
                    // K[8h..8h+7]+K[16+8h..16+8h+7] (h=lane/16) — two 16B
                    // LDS reads (weights staged per chunk).
                    const int col = nc * 16 + l;
                    const _Float16* wp =
                        sW + (kk * COUT + col) * CINPAD + kb + hi * 8;
                    v8h b0 = *reinterpret_cast<const v8h*>(wp);
                    v8h b1 = *reinterpret_cast<const v8h*>(wp + 16);
                    v16h b;
#pragma unroll
                    for (int j = 0; j < 8; ++j) { b[j] = b0[j]; b[8 + j] = b1[j]; }
                    acc[nc] = __builtin_amdgcn_wmma_f32_16x16x32_f16(
                        false, a, false, b, (short)0, acc[nc], false, false);
                }
            }
        }
    }

    // C/D layout: VGPR r -> M = r (+8 for lanes 16-31), N = lane%16.
#pragma unroll
    for (int nc = 0; nc < NC; ++nc) {
#pragma unroll
        for (int r = 0; r < 8; ++r) {
            const int M = outBase + r + hi * 8;
            if (M < nOut) {
                float v = acc[nc][r];
                if (doRelu) v = fmaxf(v, 0.0f);
                const int N = dstColOff + nc * 16 + l;
                if (dstF) {
                    dstF[(size_t)M * dstStride + N] = v;
                } else {
                    if (resid) v += (float)resid[(size_t)M * dstStride + N];
                    dstH[(size_t)M * dstStride + N] = (_Float16)v;
                    if (padZero)  // zero K-pad columns 16..31 for next stage
                        dstH[(size_t)M * dstStride + N + 16] = (_Float16)0;
                }
            }
        }
    }
}

static inline int cdiv(int a, int b) { return (a + b - 1) / b; }

extern "C" void kernel_launch(void* const* d_in, const int* in_sizes, int n_in,
                              void* d_out, int out_size, void* d_ws, size_t ws_size,
                              hipStream_t stream) {
    const float* feats    = (const float*)d_in[0];
    const int*   nbr_down = (const int*)  d_in[1];
    const int*   nbr3     = (const int*)  d_in[2];
    const float* Wd  = (const float*)d_in[3];
    const float* bd  = (const float*)d_in[4];
    const float* W00 = (const float*)d_in[5];
    const float* b00 = (const float*)d_in[6];
    const float* W01 = (const float*)d_in[7];
    const float* b01 = (const float*)d_in[8];
    const float* W10 = (const float*)d_in[9];
    const float* b10 = (const float*)d_in[10];
    const float* W11 = (const float*)d_in[11];
    const float* b11 = (const float*)d_in[12];
    const float* W12 = (const float*)d_in[13];
    const float* b12 = (const float*)d_in[14];
    const float* W2  = (const float*)d_in[15];
    const float* b2  = (const float*)d_in[16];

    const int nPts = in_sizes[0] / N_PTS_C;   // 200000
    const int nOut = in_sizes[1] / 8;         // data-dependent downsampled count

    // -------- workspace layout (f16 units, every region 32B-aligned) --------
    _Float16* ws = (_Float16*)d_ws;
    size_t off = 0;
    auto take = [&](size_t n) { _Float16* p = ws + off; off += n; return p; };
    _Float16* featsH = take((size_t)(nPts + 1) * 64);
    _Float16* hA     = take((size_t)(nOut + 1) * 64);
    _Float16* hB     = take((size_t)(nOut + 1) * 64);
    _Float16* t0     = take((size_t)(nOut + 1) * 32);
    _Float16* t1     = take((size_t)(nOut + 1) * 32);
    _Float16* t2     = take((size_t)(nOut + 1) * 32);
    _Float16* WdT    = take((size_t)8  * 64 * 64);
    _Float16* W00T   = take((size_t)81 * 16 * 64);   // [3*27, 16, 64]
    _Float16* W01T   = take((size_t)81 * 32 * 32);   // [3*27, 32, 32] (Cin 16->32)
    _Float16* W10T   = take((size_t)3  * 16 * 64);
    _Float16* W11T   = take((size_t)81 * 16 * 32);
    _Float16* W12T   = take((size_t)3  * 32 * 32);
    _Float16* W2T    = take((size_t)32 * 64 * 64);   // K padded 27 -> 32 (zeros)

    // Zero workspace: establishes padding rows, K-pad columns/slices.
    hipMemsetAsync(d_ws, 0, off * sizeof(_Float16), stream);

    // -------- precision conversion / weight transposition --------
    {
        int n = nPts * 64;
        fconv_kernel<<<cdiv(n, 256), 256, 0, stream>>>(feats, featsH, n);
    }
    auto wc = [&](const float* W, _Float16* WT, int K, int Cin, int Cout, int CinPad) {
        int total = K * Cin * Cout;
        wconv_kernel<<<cdiv(total, 256), 256, 0, stream>>>(W, WT, Cin, Cout, CinPad, total);
    };
    wc(Wd,  WdT,   8, 64, 64, 64);
    wc(W00, W00T, 81, 64, 16, 64);
    wc(W01, W01T, 81, 16, 32, 32);
    wc(W10, W10T,  3, 64, 16, 64);
    wc(W11, W11T, 81, 16, 16, 32);
    wc(W12, W12T,  3, 16, 32, 32);
    wc(W2,  W2T,  27, 64, 64, 64);

    const int nTiles = cdiv(nOut, 16);
    dim3 grid(cdiv(nTiles, 8)), block(256);
    auto shb = [](int kch, int cout, int cinPad) {
        return (size_t)kch * cout * cinPad * sizeof(_Float16);
    };

    // -------- downsample conv: h = relu(sconv(feats, nbr_down, Wd) + bd) ----
    sconv16_kernel<4, 2, 8, 8><<<grid, block, shb(8, 64, 64), stream>>>(
        featsH, 64, nbr_down, WdT, bd, nOut, nPts,
        hA, nullptr, 64, 0, nullptr, /*relu*/1, /*padZero*/0);

    // -------- 3 InceptionResNet blocks --------
    for (int i = 0; i < 3; ++i) {
        // t0 = relu(sconv(h, nbr3, W00) + b00)        [nOut, 16->pad32]
        sconv16_kernel<1, 2, 27, 27><<<grid, block, shb(27, 16, 64), stream>>>(
            hA, 64, nbr3, W00T + (size_t)i * 27 * 16 * 64,
            b00 + i * 16, nOut, nOut, t0, nullptr, 32, 0, nullptr, 1, 1);
        // t1 = relu(h @ W10 + b10)                    [nOut, 16->pad32]
        sconv16_kernel<1, 2, 1, 1><<<grid, block, shb(1, 16, 64), stream>>>(
            hA, 64, nullptr, W10T + (size_t)i * 16 * 64,
            b10 + i * 16, nOut, nOut, t1, nullptr, 32, 0, nullptr, 1, 1);
        // t2 = relu(sconv(t1, nbr3, W11) + b11)       [nOut, 16->pad32]
        sconv16_kernel<1, 1, 27, 27><<<grid, block, shb(27, 16, 32), stream>>>(
            t1, 32, nbr3, W11T + (size_t)i * 27 * 16 * 32,
            b11 + i * 16, nOut, nOut, t2, nullptr, 32, 0, nullptr, 1, 1);
        // hB[:, 0:32]  = sconv(t0, nbr3, W01) + b01 + hA[:, 0:32]
        sconv16_kernel<2, 1, 27, 27><<<grid, block, shb(27, 32, 32), stream>>>(
            t0, 32, nbr3, W01T + (size_t)i * 27 * 32 * 32,
            b01 + i * 32, nOut, nOut, hB, nullptr, 64, 0, hA, 0, 0);
        // hB[:, 32:64] = t2 @ W12 + b12 + hA[:, 32:64]
        sconv16_kernel<2, 1, 1, 1><<<grid, block, shb(1, 32, 32), stream>>>(
            t2, 32, nullptr, W12T + (size_t)i * 32 * 32,
            b12 + i * 32, nOut, nOut, hB, nullptr, 64, 32, hA, 0, 0);
        _Float16* tmp = hA; hA = hB; hB = tmp;
    }

    // -------- output head: out = sconv(h, nbr3, W2) + b2  (f32) --------
    // KOFF=27, KCH=8 -> 4 chunks (taps 27..31 are zero slices / zero rows).
    sconv16_kernel<4, 2, 27, 8><<<grid, block, shb(8, 64, 64), stream>>>(
        hA, 64, nbr3, W2T, b2, nOut, nOut,
        nullptr, (float*)d_out, 64, 0, nullptr, 0, 0);

    (void)n_in; (void)out_size; (void)ws_size;
}
